// StudentRNAFeatureExtractor_57870389347016
// MI455X (gfx1250) — compile-verified, matrix operand order checked by reading, not run on hardware
//
#include <hip/hip_runtime.h>
#include <hip/hip_bf16.h>

typedef __attribute__((ext_vector_type(16))) _Float16 v16h;
typedef __attribute__((ext_vector_type(8)))  float    v8f;

#define GTM 64
#define GTN 64
#define GTK 32
#define LDSROW 40                 // halves per LDS tile row (80 bytes -> 16B aligned chunks)
#define SM_B_BASE (64 * LDSROW)   // halves offset of B tile in shared arena

// Stage a 64x32 f16 tile into LDS. Fast path: async global->LDS B128 copies
// (contiguous 16B chunks). Slow path (OOB rows / K tail): scalar fill.
__device__ __forceinline__ void stage_tile(const _Float16* __restrict__ G, int ldg,
                                           _Float16* sm, int smBase,
                                           int row0, int rowLimit, int kk, int K, int tid) {
  const bool kfull = (kk + GTK) <= K;
  for (int idx = tid; idx < 256; idx += 128) {
    int r = idx >> 2, seg = idx & 3;              // row, 16-byte segment
    int gr = row0 + r;
    if (gr >= 0 && gr < rowLimit && kfull) {
      unsigned lds = (unsigned)((smBase + r * LDSROW + seg * 8) * 2);
      unsigned long long gaddr =
          (unsigned long long)(G + (size_t)gr * ldg + kk + seg * 8);
      asm volatile("global_load_async_to_lds_b128 %0, %1, off"
                   :: "v"(lds), "v"(gaddr) : "memory");
    } else {
#pragma unroll
      for (int j = 0; j < 8; ++j) {
        int gc = kk + seg * 8 + j;
        _Float16 v = (_Float16)0.0f;
        if (gr >= 0 && gr < rowLimit && gc < K) v = G[(size_t)gr * ldg + gc];
        sm[smBase + r * LDSROW + seg * 8 + j] = v;
      }
    }
  }
}

// ---------------- WMMA GEMM on pre-converted f16 operands -----------------
// C[M,Nc](f32) = op(A16[M,K] @ Bt16[Nc,K]^T). A rows read at (row+rowOff),
// rows outside [0,Msrc) are zero (conv taps). flags: bit0 relu, bit1 accumulate.
__global__ __launch_bounds__(128)
void wmma_gemm16_kernel(const _Float16* __restrict__ A, const _Float16* __restrict__ Bt,
                        float* __restrict__ C, const float* __restrict__ bias,
                        int M, int K, int Nc, int Msrc, int rowOff, int flags) {
  __shared__ _Float16 smem[64 * LDSROW * 2];     // [A tile | B tile], single LDS object
  const int tid  = threadIdx.x;
  const int wave = tid >> 5;
  const int lane = tid & 31;
  const int half = lane >> 4;
  const int lr   = lane & 15;
  const int m0 = blockIdx.y * GTM;
  const int n0 = blockIdx.x * GTN;

  v8f acc[4];
#pragma unroll
  for (int i = 0; i < 4; ++i)
#pragma unroll
    for (int j = 0; j < 8; ++j) acc[i][j] = 0.0f;

  for (int kk = 0; kk < K; kk += GTK) {
    stage_tile(A,  K, smem, 0,         m0 + rowOff, Msrc, kk, K, tid);
    stage_tile(Bt, K, smem, SM_B_BASE, n0,          Nc,   kk, K, tid);
    asm volatile("s_wait_asynccnt 0" ::: "memory");
    __syncthreads();

    // A fragment: lane holds row (wave*16+lr); K striping per ISA 16-bit A layout
    const int rA = (wave * 16 + lr) * LDSROW;
    v16h afrag;
#pragma unroll
    for (int v = 0; v < 8; ++v) {
      int k0 = ((v >> 2) << 4) + (half << 3) + ((v & 3) << 1);
      afrag[2 * v]     = smem[rA + k0];
      afrag[2 * v + 1] = smem[rA + k0 + 1];
    }
#pragma unroll
    for (int nt = 0; nt < 4; ++nt) {
      const int rB = SM_B_BASE + (nt * 16 + lr) * LDSROW;
      v16h bfrag;
#pragma unroll
      for (int v = 0; v < 8; ++v) {
        int k0 = ((v >> 2) << 4) + (half << 3) + ((v & 3) << 1);
        bfrag[2 * v]     = smem[rB + k0];
        bfrag[2 * v + 1] = smem[rB + k0 + 1];
      }
      acc[nt] = __builtin_amdgcn_wmma_f32_16x16x32_f16(
          false, afrag, false, bfrag, (short)0, acc[nt], false, false);
    }
    __syncthreads();
  }

  // epilogue: C/D layout -> row = half*8 + r, col = lr
#pragma unroll
  for (int nt = 0; nt < 4; ++nt) {
    int col = n0 + nt * 16 + lr;
    if (col >= Nc) continue;
#pragma unroll
    for (int r = 0; r < 8; ++r) {
      int row = m0 + wave * 16 + half * 8 + r;
      if (row >= M) continue;
      float v = acc[nt][r];
      size_t ci = (size_t)row * Nc + col;
      if (flags & 2) v += C[ci];
      if (bias) v += bias[col];
      if (flags & 1) v = fmaxf(v, 0.0f);
      C[ci] = v;
    }
  }
}

// ---------------- conversion / staging helpers ----------------
__global__ void f32_to_f16_kernel(const float* __restrict__ in, _Float16* __restrict__ out,
                                  long long n) {
  for (long long i = (long long)blockIdx.x * blockDim.x + threadIdx.x; i < n;
       i += (long long)gridDim.x * blockDim.x) out[i] = (_Float16)in[i];
}
// Bt[n*K+k] = (f16)B[k*Nc+n]
__global__ void convert_transpose_kernel(const float* __restrict__ B, _Float16* __restrict__ Bt,
                                         int K, int Nc) {
  long long total = (long long)K * Nc;
  for (long long idx = (long long)blockIdx.x * blockDim.x + threadIdx.x; idx < total;
       idx += (long long)gridDim.x * blockDim.x) {
    int n = (int)(idx % Nc), k = (int)(idx / Nc);
    Bt[(size_t)n * K + k] = (_Float16)B[idx];
  }
}
// per-tap f16 B^T slices: Wt[((t*128)+o)*256 + i] = (f16)W[o][i][t]
__global__ void conv_w_transpose16_kernel(const float* __restrict__ W, _Float16* __restrict__ Wt,
                                          int ksz) {
  int total = 128 * 256 * ksz;
  for (int idx = blockIdx.x * blockDim.x + threadIdx.x; idx < total;
       idx += gridDim.x * blockDim.x) {
    int o = idx / (256 * ksz);
    int rem = idx % (256 * ksz);
    int i = rem / ksz, t = rem % ksz;
    Wt[((size_t)t * 128 + o) * 256 + i] = (_Float16)W[idx];
  }
}

__global__ void fill_f32_kernel(float* p, float v, long long n) {
  for (long long i = (long long)blockIdx.x * blockDim.x + threadIdx.x; i < n;
       i += (long long)gridDim.x * blockDim.x) p[i] = v;
}
__global__ void fill_i32_kernel(int* p, int v, long long n) {
  for (long long i = (long long)blockIdx.x * blockDim.x + threadIdx.x; i < n;
       i += (long long)gridDim.x * blockDim.x) p[i] = v;
}

__device__ __forceinline__ int enc_f(float f) {
  int b = __float_as_int(f);
  return b >= 0 ? b : (b ^ 0x7fffffff);
}
__device__ __forceinline__ float dec_f(int k) {
  int b = k >= 0 ? k : (k ^ 0x7fffffff);
  return __int_as_float(b);
}

// a[n,h] = sum_c h[n,h,c] * att[h,c]   (one wave32 per (n,h))
__global__ void att_scores_kernel(const float* __restrict__ h, const float* __restrict__ att,
                                  float* __restrict__ out, int n_nodes, int heads) {
  int warp = blockIdx.x * (blockDim.x >> 5) + (threadIdx.x >> 5);
  int lane = threadIdx.x & 31;
  int total = n_nodes * heads;
  if (warp >= total) return;
  int nd = warp / heads, hh = warp % heads;
  const float* hp = h + (size_t)nd * heads * 256 + (size_t)hh * 256;
  const float* ap = att + (size_t)hh * 256;
  float s = 0.0f;
  for (int c = lane; c < 256; c += 32) s += hp[c] * ap[c];
  for (int off = 16; off; off >>= 1) s += __shfl_xor(s, off, 32);
  if (lane == 0) out[warp] = s;
}

__global__ void edge_max_kernel(const float* __restrict__ as_, const float* __restrict__ ad_,
                                const int* __restrict__ src, const int* __restrict__ dst,
                                int* __restrict__ mx, int E, int n_nodes, int heads) {
  int Etot = E + n_nodes;
  for (int i = blockIdx.x * blockDim.x + threadIdx.x; i < Etot;
       i += gridDim.x * blockDim.x) {
    int s = (i < E) ? src[i] : (i - E);
    int d = (i < E) ? dst[i] : (i - E);
    for (int h = 0; h < heads; ++h) {
      float l = as_[s * heads + h] + ad_[d * heads + h];
      l = (l > 0.0f) ? l : 0.2f * l;
      atomicMax(&mx[d * heads + h], enc_f(l));
    }
  }
}

__global__ void edge_expsum_kernel(const float* __restrict__ as_, const float* __restrict__ ad_,
                                   const int* __restrict__ src, const int* __restrict__ dst,
                                   const int* __restrict__ mx, float* __restrict__ ex,
                                   float* __restrict__ den, int E, int n_nodes, int heads) {
  int Etot = E + n_nodes;
  for (int i = blockIdx.x * blockDim.x + threadIdx.x; i < Etot;
       i += gridDim.x * blockDim.x) {
    int s = (i < E) ? src[i] : (i - E);
    int d = (i < E) ? dst[i] : (i - E);
    for (int h = 0; h < heads; ++h) {
      float l = as_[s * heads + h] + ad_[d * heads + h];
      l = (l > 0.0f) ? l : 0.2f * l;
      float e = __expf(l - dec_f(mx[d * heads + h]));
      ex[(size_t)i * heads + h] = e;
      atomicAdd(&den[d * heads + h], e);
    }
  }
}

__global__ void edge_scatter_kernel(const float* __restrict__ hbuf, const float* __restrict__ ex,
                                    const float* __restrict__ den, const int* __restrict__ src,
                                    const int* __restrict__ dst, float* __restrict__ out,
                                    int E, int n_nodes, int heads) {
  long long total = (long long)(E + n_nodes) * 256;
  for (long long idx = (long long)blockIdx.x * blockDim.x + threadIdx.x; idx < total;
       idx += (long long)gridDim.x * blockDim.x) {
    int c = (int)(idx & 255);
    int i = (int)(idx >> 8);
    int s = (i < E) ? src[i] : (i - E);
    int d = (i < E) ? dst[i] : (i - E);
    for (int h = 0; h < heads; ++h) {
      float alpha = ex[(size_t)i * heads + h] / (den[d * heads + h] + 1e-16f);
      atomicAdd(&out[(size_t)d * heads * 256 + (size_t)h * 256 + c],
                alpha * hbuf[(size_t)s * heads * 256 + (size_t)h * 256 + c]);
    }
  }
}

// mean-over-heads + bias + residual + LayerNorm + relu, in-place on x[n,256]
__global__ __launch_bounds__(256)
void gat_post_kernel(const float* __restrict__ gout, const float* __restrict__ bias,
                     const float* __restrict__ gamma, const float* __restrict__ beta,
                     float* __restrict__ x, int heads) {
  __shared__ float red[256];
  int n = blockIdx.x, c = threadIdx.x;
  size_t base = (size_t)n * heads * 256;
  float s = 0.0f;
  for (int h = 0; h < heads; ++h) s += gout[base + (size_t)h * 256 + c];
  float g = s / (float)heads + bias[c] + x[(size_t)n * 256 + c];
  red[c] = g;
  __syncthreads();
  for (int off = 128; off; off >>= 1) { if (c < off) red[c] += red[c + off]; __syncthreads(); }
  float mean = red[0] * (1.0f / 256.0f);
  __syncthreads();
  float dv = g - mean;
  red[c] = dv * dv;
  __syncthreads();
  for (int off = 128; off; off >>= 1) { if (c < off) red[c] += red[c + off]; __syncthreads(); }
  float var = red[0] * (1.0f / 256.0f);
  float y = (g - mean) / sqrtf(var + 1e-5f) * gamma[c] + beta[c];
  x[(size_t)n * 256 + c] = fmaxf(y, 0.0f);
}

__global__ void conv_combine_kernel(float* __restrict__ acc, const float* __restrict__ b7,
                                    const float* __restrict__ b11, const float* __restrict__ b15,
                                    long long total) {
  for (long long idx = (long long)blockIdx.x * blockDim.x + threadIdx.x; idx < total;
       idx += (long long)gridDim.x * blockDim.x) {
    int c = (int)(idx & 127);
    acc[idx] = (acc[idx] + b7[c] + b11[c] + b15[c]) * (1.0f / 3.0f);
  }
}

__global__ __launch_bounds__(256)
void col_mean_kernel(const float* __restrict__ in, float* __restrict__ out, int rows, int cols) {
  __shared__ float red[256];
  int c = blockIdx.x, t = threadIdx.x;
  float s = 0.0f;
  for (int r = t; r < rows; r += 256) s += in[(size_t)r * cols + c];
  red[t] = s;
  __syncthreads();
  for (int off = 128; off; off >>= 1) { if (t < off) red[t] += red[t + off]; __syncthreads(); }
  if (t == 0) out[c] = red[0] / (float)rows;
}

__global__ __launch_bounds__(256)
void fusion_kernel(const float* __restrict__ gf, const float* __restrict__ cf,
                   const float* __restrict__ W1, const float* __restrict__ b1,
                   const float* __restrict__ W2, const float* __restrict__ b2,
                   float* __restrict__ out) {
  __shared__ float fus[256];
  __shared__ float t1[256];
  int o = threadIdx.x;
  fus[o] = 0.5f * (gf[o] + cf[o]);
  __syncthreads();
  float s = b1[o];
  for (int c = 0; c < 256; ++c) s += fus[c] * W1[c * 256 + o];
  t1[o] = fmaxf(s, 0.0f);
  __syncthreads();
  float s2 = b2[o];
  for (int c = 0; c < 256; ++c) s2 += t1[c] * W2[c * 256 + o];
  out[o] = s2;
}

// ---------------- launcher ----------------
extern "C" void kernel_launch(void* const* d_in, const int* in_sizes, int n_in,
                              void* d_out, int out_size, void* d_ws, size_t ws_size,
                              hipStream_t stream) {
  const int DEMB = 640;
  const float* emb = (const float*)d_in[0];
  const int* eidx = (const int*)d_in[1];
  const int Nn = in_sizes[0] / DEMB;
  const int E  = in_sizes[1] / 2;
  const int* srcp = eidx;
  const int* dstp = eidx + E;

  const float* P[40];
  for (int i = 3; i < n_in && i < 43; ++i) P[i - 3] = (const float*)d_in[i];
  const float* emb_W = P[0]; const float* emb_b = P[1];
  const float* gatW[3]  = {P[2],  P[8],  P[14]};
  const float* gatAS[3] = {P[3],  P[9],  P[15]};
  const float* gatAD[3] = {P[4],  P[10], P[16]};
  const float* gatB[3]  = {P[5],  P[11], P[17]};
  const float* normG[3] = {P[6],  P[12], P[18]};
  const float* normB[3] = {P[7],  P[13], P[19]};
  const float* convW[3] = {P[20], P[22], P[24]};
  const float* convB[3] = {P[21], P[23], P[25]};
  const float* line1W = P[26]; const float* line1b = P[27];
  const float* line2W = P[28]; const float* line2b = P[29];
  const float* fus1W  = P[30]; const float* fus1b  = P[31];
  const float* fus2W  = P[32]; const float* fus2b  = P[33];

  const int hd[3] = {4, 4, 1};
  float* out = (float*)d_out;
  float* x = out + 256;                       // running node features == 2nd output
  float* ws = (float*)d_ws;

  auto al = [](size_t v) { return (v + 63) & ~(size_t)63; };
  size_t offH  = 0;
  size_t offG  = al(offH  + (size_t)Nn * 1024);
  size_t offAS = al(offG  + (size_t)Nn * 1024);
  size_t offAD = al(offAS + (size_t)Nn * 4);
  size_t offMX = al(offAD + (size_t)Nn * 4);
  size_t offDE = al(offMX + (size_t)Nn * 4);
  size_t offEX = al(offDE + (size_t)Nn * 4);
  size_t offA16 = al(offEX + (size_t)(E + Nn) * 4);        // f16 A arena: Nn*640 halves
  size_t offB16 = al(offA16 + (size_t)Nn * 320);           // f16 B arena: 262144 halves
  size_t offWT  = al(offB16 + 131072);                     // f16 conv taps: 33*32768 halves
  size_t offSM  = al(offWT + 540672);
  float* bufH = ws + offH;
  float* bufG = ws + offG;
  float* aS   = ws + offAS;
  float* aD   = ws + offAD;
  int*   mx   = (int*)(ws + offMX);
  float* den  = ws + offDE;
  float* ex   = ws + offEX;
  _Float16* A16  = (_Float16*)(ws + offA16);
  _Float16* B16  = (_Float16*)(ws + offB16);
  _Float16* Wt16 = (_Float16*)(ws + offWT);
  float* gf   = ws + offSM;
  float* cf   = gf + 256;

  auto gemm16 = [&](const _Float16* A, const _Float16* Bt, float* C, const float* bias,
                    int M, int K, int Nc, int Msrc, int rowOff, int flags) {
    dim3 g((Nc + GTN - 1) / GTN, (M + GTM - 1) / GTM);
    wmma_gemm16_kernel<<<g, dim3(128), 0, stream>>>(A, Bt, C, bias, M, K, Nc, Msrc, rowOff, flags);
  };
  auto toH = [&](const float* src, _Float16* dst, long long n) {
    int blk = (int)((n + 255) / 256); if (blk > 65535) blk = 65535;
    f32_to_f16_kernel<<<blk, 256, 0, stream>>>(src, dst, n);
  };
  auto convT = [&](const float* B, _Float16* Bt, int K, int Nc) {
    long long n = (long long)K * Nc;
    int blk = (int)((n + 255) / 256); if (blk > 65535) blk = 65535;
    convert_transpose_kernel<<<blk, 256, 0, stream>>>(B, Bt, K, Nc);
  };
  auto fillf = [&](float* p, float v, long long n) {
    int blk = (int)((n + 255) / 256); if (blk > 65535) blk = 65535;
    fill_f32_kernel<<<blk, 256, 0, stream>>>(p, v, n);
  };
  auto filli = [&](int* p, int v, long long n) {
    int blk = (int)((n + 255) / 256); if (blk > 65535) blk = 65535;
    fill_i32_kernel<<<blk, 256, 0, stream>>>(p, v, n);
  };

  const int negInfEnc = (int)(0xFF800000u ^ 0x7FFFFFFFu);  // enc(-inf)

  // 1. x = relu(emb @ emb_W + emb_b)
  toH(emb, A16, (long long)Nn * DEMB);
  convT(emb_W, B16, DEMB, 256);
  gemm16(A16, B16, x, emb_b, Nn, DEMB, 256, Nn, 0, 1);

  // 2. three GAT layers
  for (int L = 0; L < 3; ++L) {
    int h = hd[L];
    toH(x, A16, (long long)Nn * 256);
    convT(gatW[L], B16, 256, h * 256);
    gemm16(A16, B16, bufH, nullptr, Nn, 256, h * 256, Nn, 0, 0);
    int pairs = Nn * h;
    att_scores_kernel<<<(pairs + 3) / 4, 128, 0, stream>>>(bufH, gatAS[L], aS, Nn, h);
    att_scores_kernel<<<(pairs + 3) / 4, 128, 0, stream>>>(bufH, gatAD[L], aD, Nn, h);
    filli(mx, negInfEnc, (long long)Nn * h);
    fillf(den, 0.0f, (long long)Nn * h);
    int eb = (E + Nn + 255) / 256; if (eb > 65535) eb = 65535;
    edge_max_kernel<<<eb, 256, 0, stream>>>(aS, aD, srcp, dstp, mx, E, Nn, h);
    edge_expsum_kernel<<<eb, 256, 0, stream>>>(aS, aD, srcp, dstp, mx, ex, den, E, Nn, h);
    fillf(bufG, 0.0f, (long long)Nn * h * 256);
    long long st = (long long)(E + Nn) * 256;
    int sb = (int)((st + 255) / 256); if (sb > 131072) sb = 131072;
    edge_scatter_kernel<<<sb, 256, 0, stream>>>(bufH, ex, den, srcp, dstp, bufG, E, Nn, h);
    gat_post_kernel<<<Nn, 256, 0, stream>>>(bufG, gatB[L], normG[L], normB[L], x, h);
  }

  // 3. convs as per-tap shifted accumulating GEMMs (f16 B^T tap slices)
  const int ksz[3] = {7, 11, 15};
  const int tapBase[3] = {0, 7, 18};
  for (int j = 0; j < 3; ++j)
    conv_w_transpose16_kernel<<<256, 256, 0, stream>>>(
        convW[j], Wt16 + (size_t)tapBase[j] * 128 * 256, ksz[j]);
  float* convacc = bufG;                       // [Nn,128]
  toH(x, A16, (long long)Nn * 256);
  fillf(convacc, 0.0f, (long long)Nn * 128);
  for (int j = 0; j < 3; ++j) {
    int pad = ksz[j] / 2;
    for (int t = 0; t < ksz[j]; ++t)
      gemm16(A16, Wt16 + (size_t)(tapBase[j] + t) * 128 * 256, convacc, nullptr,
             Nn, 256, 128, Nn, t - pad, 2 /*accumulate*/);
  }
  {
    long long tot = (long long)Nn * 128;
    int blk = (int)((tot + 255) / 256); if (blk > 65535) blk = 65535;
    conv_combine_kernel<<<blk, 256, 0, stream>>>(convacc, convB[0], convB[1], convB[2], tot);
  }

  // 4. line1 (relu) -> line2
  toH(convacc, A16, (long long)Nn * 128);
  convT(line1W, B16, 128, 512);
  float* line1out = bufH;                      // [Nn,512]
  gemm16(A16, B16, line1out, line1b, Nn, 128, 512, Nn, 0, 1);
  toH(line1out, A16, (long long)Nn * 512);
  convT(line2W, B16, 512, 256);
  float* line2out = bufG;                      // [Nn,256]
  gemm16(A16, B16, line2out, line2b, Nn, 512, 256, Nn, 0, 0);

  // 5. pooling + fusion MLP -> enhanced (d_out[0:256])
  col_mean_kernel<<<256, 256, 0, stream>>>(x, gf, Nn, 256);
  col_mean_kernel<<<256, 256, 0, stream>>>(line2out, cf, Nn, 256);
  fusion_kernel<<<1, 256, 0, stream>>>(gf, cf, fus1W, fus1b, fus2W, fus2b, out);
  (void)ws_size; (void)out_size; (void)n_in;
}
